// ModConv1x1_21354577395714
// MI455X (gfx1250) — matrix operand order repaired and checked by reading, not compiled, not run on hardware
//
#include <hip/hip_runtime.h>

// out[b,o,p] = sum_c (weight[o,c]*style[b,c]) * x[b,c,p]
// B=16, CIN=512, COUT=128, HW=4096. f16 WMMA, f32 accumulate.
//
// Roofline: 8.6 GFLOP vs ~160 MB HBM traffic -> ~51 FLOP/B; at 23.3 TB/s we
// need >1.2 PFLOPS to stay memory-bound. f16 16x16x32 WMMA clears that;
// f32 16x16x4 WMMA would not. So: convert to f16 in registers during LDS
// staging, accumulate in f32.

typedef _Float16 v4h  __attribute__((ext_vector_type(4)));
typedef _Float16 v8h  __attribute__((ext_vector_type(8)));
typedef _Float16 v16h __attribute__((ext_vector_type(16)));
typedef float    v4f  __attribute__((ext_vector_type(4)));
typedef float    v8f  __attribute__((ext_vector_type(8)));

#define CIN     512
#define COUT    128
#define HW      4096
#define NB      16
#define B_N     128            // pixels per workgroup tile
#define T_K     32             // K per stage (one WMMA K-slab)
#define KITERS  (CIN / T_K)    // 16
#define ASTRIDE 40             // halves per weight row (80 B, 16B-aligned, conflict-free frag reads)
#define XSTRIDE 56             // halves per pixel row (112 B, 16B-aligned frag reads)

union F16x16 { v16h v; v8h h[2]; };

__global__ __launch_bounds__(256)
void ModConv1x1_21354577395714_kernel(const float* __restrict__ x,
                                      const float* __restrict__ style,
                                      const float* __restrict__ weight,
                                      float* __restrict__ out) {
  __shared__ _Float16 shA[2][COUT][ASTRIDE];   // modulated weight chunk, [o][k] f16
  __shared__ _Float16 shX[2][B_N][XSTRIDE];    // x chunk transposed, [pixel][k] f16

  const int tid  = threadIdx.x;
  const int lane = tid & 31;
  const int wv   = tid >> 5;     // wave id 0..7 -> M tile
  const int lg   = lane >> 4;    // half-wave group (K sub-range select)
  const int lr   = lane & 15;

  const int b  = blockIdx.y;
  const int p0 = blockIdx.x * B_N;

  const float* xb = x + (size_t)b * CIN * HW;
  const float* sb = style + b * CIN;

  v8f acc[8] = {};

  auto stage = [&](int kk, int buf) {
    const int c0 = kk * T_K;
    // ---- modulated weight chunk: 128 rows x 32 cols = 1024 float4 tasks
    #pragma unroll
    for (int j = 0; j < 4; ++j) {
      const int f  = tid + j * 256;
      const int o  = f >> 3;             // 8 float4 per 32-col row
      const int cc = (f & 7) << 2;
      const v4f w4 = *(const v4f*)(weight + o * CIN + c0 + cc);
      const v4f s4 = *(const v4f*)(sb + c0 + cc);
      v4h h;
      h.x = (_Float16)(w4.x * s4.x);
      h.y = (_Float16)(w4.y * s4.y);
      h.z = (_Float16)(w4.z * s4.z);
      h.w = (_Float16)(w4.w * s4.w);
      *(v4h*)&shA[buf][o][cc] = h;
    }
    // ---- x chunk, transposed into [pixel][k]: 128 pixels x 8 k-quads
    #pragma unroll
    for (int j = 0; j < 4; ++j) {
      const int f  = tid + j * 256;
      const int p  = f & (B_N - 1);
      const int c4 = f >> 7;             // 0..7 -> k = 4*c4..4*c4+3
      const float* src = xb + (size_t)(c0 + (c4 << 2)) * HW + (p0 + p);
      v4h h;
      h.x = (_Float16)src[0];
      h.y = (_Float16)src[HW];
      h.z = (_Float16)src[2 * HW];
      h.w = (_Float16)src[3 * HW];
      *(v4h*)&shX[buf][p][c4 << 2] = h;
    }
  };

  stage(0, 0);

  #pragma unroll 1
  for (int kk = 0; kk < KITERS; ++kk) {
    const int buf = kk & 1;
    __syncthreads();                     // stage(buf) visible; prev reads of buf^1 done
    if (kk + 1 < KITERS) stage(kk + 1, buf ^ 1);

    // A fragment: 16x32 f16. group0 lanes: K0..7 (v8h 0) + K16..23 (v8h 2);
    // group1 lanes: K8..15 (v8h 1) + K24..31 (v8h 3).
    F16x16 a;
    const v8h* arow = (const v8h*)&shA[buf][wv * 16 + lr][0];
    a.h[0] = arow[lg];
    a.h[1] = arow[lg + 2];

    // Preload ALL 8 B fragments so the ds_loads issue as one batch.
    F16x16 bf[8];
    #pragma unroll
    for (int nt = 0; nt < 8; ++nt) {
      // B fragment: 32x16 f16, lane = column N. group0: K0..7 + K16..23, group1: K8..15 + K24..31.
      const v8h* xrow = (const v8h*)&shX[buf][nt * 16 + lr][0];
      bf[nt].h[0] = xrow[lg];
      bf[nt].h[1] = xrow[lg + 2];
    }

    // Hard scheduling fence: forbid the backend from sinking the ds_loads
    // into the WMMA sequence (it otherwise coalesces bf[] into one register
    // octet and re-serializes ds_load -> s_wait_dscnt -> wmma per tile).
    // Result: one dscnt drain, then 8 back-to-back WMMAs (independent
    // accumulators -> no D->A/B hazard, good XDL pipelining).
    __builtin_amdgcn_sched_barrier(0);

    #pragma unroll
    for (int nt = 0; nt < 8; ++nt) {
      acc[nt] = __builtin_amdgcn_wmma_f32_16x16x32_f16(
          false, a.v, false, bf[nt].v, (short)0, acc[nt], false, false);
    }
  }

  // Epilogue: C layout — VGPR r: lanes 0..15 -> M=r, lanes 16..31 -> M=r+8; N = lane%16
  float* ob = out + (size_t)b * COUT * HW + p0;
  #pragma unroll
  for (int nt = 0; nt < 8; ++nt) {
    const int n = nt * 16 + lr;
    #pragma unroll
    for (int r = 0; r < 8; ++r) {
      const int m = wv * 16 + lg * 8 + r;
      ob[(size_t)m * HW + n] = acc[nt][r];
    }
  }
}

extern "C" void kernel_launch(void* const* d_in, const int* in_sizes, int n_in,
                              void* d_out, int out_size, void* d_ws, size_t ws_size,
                              hipStream_t stream) {
  (void)in_sizes; (void)n_in; (void)out_size; (void)d_ws; (void)ws_size;
  const float* x      = (const float*)d_in[0];
  const float* style  = (const float*)d_in[1];
  const float* weight = (const float*)d_in[2];
  float* out = (float*)d_out;

  dim3 grid(HW / B_N, NB, 1);   // 32 pixel tiles x 16 batches = 512 workgroups
  ModConv1x1_21354577395714_kernel<<<grid, 256, 0, stream>>>(x, style, weight, out);
}
